// Encoder_68195490726471
// MI455X (gfx1250) — matrix operand order repaired and checked by reading, not compiled, not run on hardware
//
#include <hip/hip_runtime.h>
#include <math.h>

// ---------------- problem constants (match reference setup_inputs) ----------
#define Bsz  64
#define S5c  256
#define Dm   256     // model dim
#define Hh   8       // heads
#define HDc  32      // head dim
#define DFFc 1024
#define T3c  64      // patches per traj, stage 3
#define T2c  16      // patches per traj, stage 2
#define NP3  (Bsz * T3c)   // 4096
#define NP2  (Bsz * T2c)   // 1024
#define Lc   2

// GEMM tiling
#define MBLK 128
#define NBLK 64
#define KSTEP 32
#define LSTR 36      // LDS row stride in floats (32 + 4 pad -> conflict-free frags)

typedef __attribute__((ext_vector_type(16))) __bf16 v16bf;
typedef __attribute__((ext_vector_type(8)))  float  v8f;
typedef __attribute__((ext_vector_type(4)))  unsigned int u32x4;
typedef __attribute__((ext_vector_type(8)))  int i32x8;
typedef __attribute__((ext_vector_type(4)))  int i32x4;

// ---------------- device helpers -------------------------------------------
__device__ __forceinline__ float pe_val(int pos, int d) {
  float div = __expf((float)(d & ~1) * (-9.210340371976184f / (float)Dm));
  float ang = (float)pos * div;
  return (d & 1) ? __cosf(ang) : __sinf(ang);
}

__device__ __forceinline__ __bf16 to_bf16(float f) { return (__bf16)f; }

__device__ __forceinline__ v16bf pack16(float4 f0, float4 f1, float4 f2, float4 f3) {
  v16bf v;
  v[0]  = to_bf16(f0.x); v[1]  = to_bf16(f0.y); v[2]  = to_bf16(f0.z); v[3]  = to_bf16(f0.w);
  v[4]  = to_bf16(f1.x); v[5]  = to_bf16(f1.y); v[6]  = to_bf16(f1.z); v[7]  = to_bf16(f1.w);
  v[8]  = to_bf16(f2.x); v[9]  = to_bf16(f2.y); v[10] = to_bf16(f2.z); v[11] = to_bf16(f2.w);
  v[12] = to_bf16(f3.x); v[13] = to_bf16(f3.y); v[14] = to_bf16(f3.z); v[15] = to_bf16(f3.w);
  return v;
}

__device__ __forceinline__ v8f wmma_bf16(v16bf a, v16bf b, v8f c) {
  return __builtin_amdgcn_wmma_f32_16x16x32_bf16(false, a, false, b, (short)0, c, false, false);
}

// Flat LDS address low 32 bits == byte offset within the workgroup's LDS.
__device__ __forceinline__ unsigned lds_off_of(const void* p) {
  return (unsigned)(size_t)p;
}

// Tensor Data Mover: DMA a 2D f32 tile [rows x KSTEP] (row stride = strideElems)
// from global memory into LDS at lds_off, inserting a 4-DWORD pad after every
// 32 DWORDs so LDS rows land at stride LSTR floats.
__device__ __forceinline__ void tdm_load_tile(const float* gsrc, unsigned lds_off,
                                              int rows, int strideElems) {
  unsigned long long ga = (unsigned long long)(size_t)gsrc;
  u32x4 g0;
  g0[0] = 1u;                                   // count=1 (valid user descriptor)
  g0[1] = lds_off;                              // lds_addr (bytes)
  g0[2] = (unsigned)(ga & 0xFFFFFFFFu);         // global_addr[31:0]
  g0[3] = (unsigned)((ga >> 32) & 0x01FFFFFFu)  // global_addr[56:32]
          | (2u << 30);                         // type = 2 ("image")
  // group1: data_size=4B (code 2); pad_enable, pad_interval=32 DWORDs (code 4),
  // pad_amount=4 DWORDs (code 3); dims/tile dims/stride for the 2D tile.
  unsigned w0  = (2u << 16) | (1u << 20) | (4u << 22) | (3u << 25);
  unsigned td0 = (unsigned)KSTEP;               // tensor_dim0 (elements)
  unsigned td1 = (unsigned)rows;                // tensor_dim1
  unsigned long long st = (unsigned long long)strideElems;
  i32x8 g1;
  g1[0] = (int)w0;
  g1[1] = (int)((td0 & 0xFFFFu) << 16);                       // [63:48] td0 lo
  g1[2] = (int)((td0 >> 16) | ((td1 & 0xFFFFu) << 16));       // td0 hi | td1 lo
  g1[3] = (int)((td1 >> 16) | ((unsigned)KSTEP << 16));       // td1 hi | tile_dim0
  g1[4] = (int)(unsigned)rows;                                // tile_dim1 (tile_dim2=0)
  g1[5] = (int)(st & 0xFFFFFFFFu);                            // dim0 stride lo
  g1[6] = (int)((st >> 32) & 0xFFFFu);                        // dim0 stride hi
  g1[7] = 0;
  i32x4 z4 = {0, 0, 0, 0};
  i32x8 z8 = {0, 0, 0, 0, 0, 0, 0, 0};
  __builtin_amdgcn_tensor_load_to_lds(g0, g1, z4, z4, z8, 0);
}

// ---------------- offsets (exclusive scans of ragged lengths) --------------
__global__ void offsets_kernel(const int* __restrict__ t5, const int* __restrict__ p3,
                               const int* __restrict__ t3, const int* __restrict__ p2,
                               const int* __restrict__ t2,
                               int* toff5, int* poff3, int* toff3, int* poff2, int* toff2) {
  if (threadIdx.x != 0 || blockIdx.x != 0) return;
  int a = 0; toff5[0] = 0; for (int i = 0; i < Bsz; i++) { a += t5[i]; toff5[i + 1] = a; }
  a = 0; poff3[0] = 0; for (int i = 0; i < NP3; i++) { a += p3[i]; poff3[i + 1] = a; }
  a = 0; toff3[0] = 0; for (int i = 0; i < Bsz; i++) { a += t3[i]; toff3[i + 1] = a; }
  a = 0; poff2[0] = 0; for (int i = 0; i < NP2; i++) { a += p2[i]; poff2[i + 1] = a; }
  a = 0; toff2[0] = 0; for (int i = 0; i < Bsz; i++) { a += t2[i]; toff2[i + 1] = a; }
}

// ---------------- elementwise: x = in + PE ---------------------------------
__global__ void add_pe_kernel(float* __restrict__ out, const float* __restrict__ in, int S) {
  int row = blockIdx.x;
  int s = row % S;
  int t = threadIdx.x;
  size_t idx = (size_t)row * Dm + t;
  out[idx] = in[idx] + pe_val(s, t);
}

__global__ void scatter_pe_kernel(float* __restrict__ out, const float* __restrict__ xv,
                                  const int* __restrict__ toff, const int* __restrict__ tlen,
                                  int S) {
  int row = blockIdx.x;
  int b = row / S, s = row % S, t = threadIdx.x;
  float v = 0.f;
  if (s < tlen[b]) v = xv[(size_t)(toff[b] + s) * Dm + t];
  out[(size_t)row * Dm + t] = v + pe_val(s, t);
}

// ---------------- WMMA GEMM: C[M,N] = A[M,K] * B[N,K]^T + bias, opt ReLU ----
// Workgroup tile 128x64 (8 waves, each 32x32 = 2x2 WMMA accumulators).
// A/B tiles staged in LDS by the Tensor Data Mover, double-buffered.
// Requires: M % 128 == 0, N % 64 == 0, K % 32 == 0 (true for all call sites).
__global__ void __launch_bounds__(256)
gemm_nt_kernel(const float* __restrict__ A, int lda,
               const float* __restrict__ Bm, int ldb,
               float* __restrict__ C, int ldc,
               int M, int N, int K,
               const float* __restrict__ bias, int relu) {
  __shared__ float sA[2][MBLK * LSTR];   // 2 x 18 KB
  __shared__ float sB[2][NBLK * LSTR];   // 2 x  9 KB
  int wave = threadIdx.x >> 5;
  int lane = threadIdx.x & 31;
  int nTilesX = N / NBLK;
  int m0 = (blockIdx.x / nTilesX) * MBLK;
  int n0 = (blockIdx.x % nTilesX) * NBLK;
  int wr = wave >> 1, wc = wave & 1;     // 4x2 wave grid over 128x64
  int r15 = lane & 15, hl = lane >> 4;

  v8f acc[2][2];
#pragma unroll
  for (int i = 0; i < 2; i++)
#pragma unroll
    for (int j = 0; j < 2; j++)
      acc[i][j] = (v8f){0.f, 0.f, 0.f, 0.f, 0.f, 0.f, 0.f, 0.f};

  int nk = K / KSTEP;
  if (wave == 0) {
    tdm_load_tile(A  + (size_t)m0 * lda, lds_off_of(&sA[0][0]), MBLK, lda);
    tdm_load_tile(Bm + (size_t)n0 * ldb, lds_off_of(&sB[0][0]), NBLK, ldb);
  }

  for (int kt = 0; kt < nk; kt++) {
    int cur = kt & 1;
    if (wave == 0) {
      if (kt + 1 < nk) {
        int kb = (kt + 1) * KSTEP;
        tdm_load_tile(A  + (size_t)m0 * lda + kb, lds_off_of(&sA[1 - cur][0]), MBLK, lda);
        tdm_load_tile(Bm + (size_t)n0 * ldb + kb, lds_off_of(&sB[1 - cur][0]), NBLK, ldb);
        __builtin_amdgcn_s_wait_tensorcnt(2);   // current pair complete (in-order)
      } else {
        __builtin_amdgcn_s_wait_tensorcnt(0);
      }
      asm volatile("" ::: "memory");
    }
    __syncthreads();

    v16bf afrag[2], bfrag[2];
#pragma unroll
    for (int mi = 0; mi < 2; mi++) {
      const float* ar = &sA[cur][(wr * 32 + mi * 16 + r15) * LSTR];
      float4 a0 = *(const float4*)(ar + hl * 8);
      float4 a1 = *(const float4*)(ar + hl * 8 + 4);
      float4 a2 = *(const float4*)(ar + 16 + hl * 8);
      float4 a3 = *(const float4*)(ar + 16 + hl * 8 + 4);
      afrag[mi] = pack16(a0, a1, a2, a3);
    }
#pragma unroll
    for (int ni = 0; ni < 2; ni++) {
      const float* br = &sB[cur][(wc * 32 + ni * 16 + r15) * LSTR];
      float4 b0 = *(const float4*)(br + hl * 8);
      float4 b1 = *(const float4*)(br + hl * 8 + 4);
      float4 b2 = *(const float4*)(br + 16 + hl * 8);
      float4 b3 = *(const float4*)(br + 16 + hl * 8 + 4);
      bfrag[ni] = pack16(b0, b1, b2, b3);
    }
    acc[0][0] = wmma_bf16(afrag[0], bfrag[0], acc[0][0]);
    acc[0][1] = wmma_bf16(afrag[0], bfrag[1], acc[0][1]);
    acc[1][0] = wmma_bf16(afrag[1], bfrag[0], acc[1][0]);
    acc[1][1] = wmma_bf16(afrag[1], bfrag[1], acc[1][1]);
    __syncthreads();
  }

#pragma unroll
  for (int mi = 0; mi < 2; mi++) {
#pragma unroll
    for (int ni = 0; ni < 2; ni++) {
      int mb = m0 + wr * 32 + mi * 16;
      int n = n0 + wc * 32 + ni * 16 + r15;   // C/D: lane N = lane % 16
      float bia = bias ? bias[n] : 0.f;
#pragma unroll
      for (int r = 0; r < 8; r++) {           // VGPR r -> M = r + 8*(lane/16)
        float v = acc[mi][ni][r] + bia;
        if (relu) v = v > 0.f ? v : 0.f;
        C[(size_t)(mb + r + 8 * hl) * ldc + n] = v;
      }
    }
  }
}

// ---------------- fused attention (online softmax, K/V in LDS) -------------
__global__ void attn_kernel(const float* __restrict__ qkv, float* __restrict__ out,
                            const int* __restrict__ lens, int S, float scale) {
  __shared__ float sK[S5c * HDc];   // 32 KB
  __shared__ float sV[S5c * HDc];   // 32 KB
  int b = blockIdx.x / Hh, h = blockIdx.x % Hh;
  int tid = threadIdx.x;
  for (int i = tid; i < S * HDc; i += 256) {
    int s = i / HDc, j = i % HDc;
    size_t base = (size_t)(b * S + s) * (3 * Dm) + h * HDc + j;
    sK[i] = qkv[base + Dm];
    sV[i] = qkv[base + 2 * Dm];
  }
  __syncthreads();
  if (tid >= S) return;
  int q = tid;
  int klen = lens[b]; if (klen > S) klen = S;
  float Q[HDc];
  size_t qbase = (size_t)(b * S + q) * (3 * Dm) + h * HDc;
#pragma unroll
  for (int j = 0; j < HDc; j++) Q[j] = qkv[qbase + j];
  float m = -3.0e38f, l = 0.f, acc[HDc];
#pragma unroll
  for (int j = 0; j < HDc; j++) acc[j] = 0.f;
  for (int k = 0; k < klen; k++) {
    float s = 0.f;
#pragma unroll
    for (int j = 0; j < HDc; j++) s += Q[j] * sK[k * HDc + j];
    s *= scale;
    float nm = s > m ? s : m;
    float corr = __expf(m - nm);
    float p = __expf(s - nm);
    l = l * corr + p;
#pragma unroll
    for (int j = 0; j < HDc; j++) acc[j] = acc[j] * corr + p * sV[k * HDc + j];
    m = nm;
  }
  float inv = l > 0.f ? 1.f / l : 0.f;
  size_t obase = (size_t)(b * S + q) * Dm + h * HDc;
#pragma unroll
  for (int j = 0; j < HDc; j++) out[obase + j] = acc[j] * inv;
}

// ---------------- residual + LayerNorm (D == 256 == blockDim) --------------
__global__ void ln_kernel(float* __restrict__ out, const float* __restrict__ a,
                          const float* __restrict__ delta,
                          const float* __restrict__ g, const float* __restrict__ bb) {
  __shared__ float red[256];
  int row = blockIdx.x, t = threadIdx.x;
  size_t idx = (size_t)row * Dm + t;
  float v = a[idx] + (delta ? delta[idx] : 0.f);
  red[t] = v; __syncthreads();
  for (int o = 128; o > 0; o >>= 1) { if (t < o) red[t] += red[t + o]; __syncthreads(); }
  float mean = red[0] * (1.f / Dm); __syncthreads();
  float c = v - mean;
  red[t] = c * c; __syncthreads();
  for (int o = 128; o > 0; o >>= 1) { if (t < o) red[t] += red[t + o]; __syncthreads(); }
  float var = red[0] * (1.f / Dm);
  out[idx] = c * rsqrtf(var + 1e-5f) * g[t] + bb[t];
}

// ---------------- ragged gather: padded [Bx,S,D] -> flat [total,D] ---------
__global__ void flatten_kernel(float* __restrict__ dst, const float* __restrict__ src,
                               const int* __restrict__ toff, int Bx, int S) {
  __shared__ int sb;
  int row = blockIdx.x;
  if (row >= toff[Bx]) return;           // uniform
  if (threadIdx.x == 0) {
    int b = 0;
    while (b + 1 < Bx && toff[b + 1] <= row) b++;
    sb = b;
  }
  __syncthreads();
  int b = sb, pos = row - toff[b];
  dst[(size_t)row * Dm + threadIdx.x] = src[(size_t)(b * S + pos) * Dm + threadIdx.x];
}

// ---------------- pool: per-token score = relu(h) . W2 + b2 ----------------
__global__ void pool_score_kernel(float* __restrict__ scores, const float* __restrict__ hbuf,
                                  const float* __restrict__ W2, const float* __restrict__ b2) {
  __shared__ float red[256];
  int row = blockIdx.x, t = threadIdx.x;
  float v = hbuf[(size_t)row * Dm + t];
  v = v > 0.f ? v : 0.f;
  red[t] = v * W2[t]; __syncthreads();
  for (int o = 128; o > 0; o >>= 1) { if (t < o) red[t] += red[t + o]; __syncthreads(); }
  if (t == 0) scores[row] = red[0] + b2[0];
}

// ---------------- pool: softmax over patch slots + weighted sum ------------
__global__ void pool_combine_kernel(float* __restrict__ outv, const float* __restrict__ flat,
                                    const float* __restrict__ scores,
                                    const int* __restrict__ poff) {
  int p = blockIdx.x, t = threadIdx.x;
  int o = poff[p], len = poff[p + 1] - o;
  float m = -3.0e38f;
  for (int i = 0; i < len; i++) { float s = scores[o + i]; m = s > m ? s : m; }
  float denom = 0.f;
  for (int i = 0; i < len; i++) denom += __expf(scores[o + i] - m);
  float inv = denom > 0.f ? 1.f / denom : 0.f;
  float acc = 0.f;
  for (int i = 0; i < len; i++)
    acc += __expf(scores[o + i] - m) * inv * flat[(size_t)(o + i) * Dm + t];
  outv[(size_t)p * Dm + t] = acc;
}

// ---------------- host-side orchestration ----------------------------------
static inline void launch_gemm(const float* A, int lda, const float* Bm, int ldb,
                               float* C, int ldc, int M, int N, int K,
                               const float* bias, bool relu, hipStream_t s) {
  int blocks = (M / MBLK) * (N / NBLK);
  gemm_nt_kernel<<<blocks, 256, 0, s>>>(A, lda, Bm, ldb, C, ldc, M, N, K, bias, relu ? 1 : 0);
}

static void run_encoder(float* x, int Bx, int S, const float* const* P,
                        const int* lens, float* bufA, float* bufB, float* bufC,
                        hipStream_t st) {
  int T = Bx * S;
  const float scale = 0.17677669529663687f;  // 1/sqrt(32)
  for (int li = 0; li < Lc; li++) {
    const float* Wqkv = P[0]  + (size_t)li * 3 * Dm * Dm;
    const float* bqkv = P[1]  + (size_t)li * 3 * Dm;
    const float* Wo   = P[2]  + (size_t)li * Dm * Dm;
    const float* bo   = P[3]  + (size_t)li * Dm;
    const float* l1g  = P[4]  + (size_t)li * Dm;
    const float* l1b  = P[5]  + (size_t)li * Dm;
    const float* W1   = P[6]  + (size_t)li * DFFc * Dm;
    const float* b1   = P[7]  + (size_t)li * DFFc;
    const float* W2   = P[8]  + (size_t)li * Dm * DFFc;
    const float* b2   = P[9]  + (size_t)li * Dm;
    const float* l2g  = P[10] + (size_t)li * Dm;
    const float* l2b  = P[11] + (size_t)li * Dm;

    launch_gemm(x, Dm, Wqkv, Dm, bufA, 3 * Dm, T, 3 * Dm, Dm, bqkv, false, st);   // QKV
    attn_kernel<<<Bx * Hh, 256, 0, st>>>(bufA, bufB, lens, S, scale);             // SDPA
    launch_gemm(bufB, Dm, Wo, Dm, bufC, Dm, T, Dm, Dm, bo, false, st);            // out proj
    ln_kernel<<<T, 256, 0, st>>>(x, x, bufC, l1g, l1b);                           // LN(x+o)
    launch_gemm(x, Dm, W1, Dm, bufA, DFFc, T, DFFc, Dm, b1, true, st);            // FFN1+ReLU
    launch_gemm(bufA, DFFc, W2, DFFc, bufB, Dm, T, Dm, DFFc, b2, false, st);      // FFN2
    ln_kernel<<<T, 256, 0, st>>>(x, x, bufB, l2g, l2b);                           // LN(x+f)
  }
}

static void run_pool(const float* x, int Bx, int S, const int* toff, const int* poff,
                     int nPatch, const float* const* PP, float* xv,
                     float* bufA, float* bufC, float* scbuf, hipStream_t st) {
  int maxRows = Bx * S;
  flatten_kernel<<<maxRows, 256, 0, st>>>(bufC, x, toff, Bx, S);
  launch_gemm(bufC, Dm, PP[0], Dm, bufA, Dm, maxRows, Dm, Dm, PP[1], false, st);  // W1+b1
  ln_kernel<<<maxRows, 256, 0, st>>>(bufA, bufA, nullptr, PP[2], PP[3]);          // LN
  pool_score_kernel<<<maxRows, 256, 0, st>>>(scbuf, bufA, PP[4], PP[5]);          // relu->W2
  pool_combine_kernel<<<nPatch, 256, 0, st>>>(xv, bufC, scbuf, poff);             // softmax+sum
}

extern "C" void kernel_launch(void* const* d_in, const int* in_sizes, int n_in,
                              void* d_out, int out_size, void* d_ws, size_t ws_size,
                              hipStream_t stream) {
  (void)in_sizes; (void)n_in; (void)out_size; (void)ws_size;
  const float* x_in = (const float*)d_in[0];
  const float* ps5[12]; for (int i = 0; i < 12; i++) ps5[i] = (const float*)d_in[1 + i];
  const float* ps3[12]; for (int i = 0; i < 12; i++) ps3[i] = (const float*)d_in[13 + i];
  const float* ps2[12]; for (int i = 0; i < 12; i++) ps2[i] = (const float*)d_in[25 + i];
  const float* pool3[6]; for (int i = 0; i < 6; i++) pool3[i] = (const float*)d_in[37 + i];
  const float* pool2[6]; for (int i = 0; i < 6; i++) pool2[i] = (const float*)d_in[43 + i];
  const int* t5 = (const int*)d_in[49];
  const int* t3 = (const int*)d_in[50];
  const int* p3 = (const int*)d_in[51];
  const int* t2 = (const int*)d_in[52];
  const int* p2 = (const int*)d_in[53];

  // outputs: x_s5, x_s3, x_s2 concatenated
  float* out5 = (float*)d_out;
  float* out3 = out5 + (size_t)Bsz * S5c * Dm;
  float* out2 = out3 + (size_t)Bsz * T3c * Dm;

  // workspace layout (floats)
  float* ws = (float*)d_ws;
  float* bufA = ws;                                    // 16,777,216 (qkv / ffn1)
  float* bufB = bufA + (size_t)Bsz * S5c * DFFc;       //  4,194,304
  float* bufC = bufB + (size_t)Bsz * S5c * Dm;         //  4,194,304
  float* scbuf = bufC + (size_t)Bsz * S5c * Dm;        //     16,384
  float* xv3 = scbuf + (size_t)Bsz * S5c;              //  1,048,576
  float* xv2 = xv3 + (size_t)NP3 * Dm;                 //    262,144
  int* ioff  = (int*)(xv2 + (size_t)NP2 * Dm);
  int* toff5 = ioff;
  int* poff3 = toff5 + (Bsz + 1);
  int* toff3 = poff3 + (NP3 + 1);
  int* poff2 = toff3 + (Bsz + 1);
  int* toff2 = poff2 + (NP2 + 1);

  offsets_kernel<<<1, 1, 0, stream>>>(t5, p3, t3, p2, t2, toff5, poff3, toff3, poff2, toff2);

  // ---- stage s5 ----
  add_pe_kernel<<<Bsz * S5c, 256, 0, stream>>>(out5, x_in, S5c);
  run_encoder(out5, Bsz, S5c, ps5, t5, bufA, bufB, bufC, stream);

  // ---- pool s3, stage s3 ----
  run_pool(out5, Bsz, S5c, toff5, poff3, NP3, pool3, xv3, bufA, bufC, scbuf, stream);
  scatter_pe_kernel<<<Bsz * T3c, 256, 0, stream>>>(out3, xv3, toff3, t3, T3c);
  run_encoder(out3, Bsz, T3c, ps3, t3, bufA, bufB, bufC, stream);

  // ---- pool s2, stage s2 ----
  run_pool(out3, Bsz, T3c, toff3, poff2, NP2, pool2, xv2, bufA, bufC, scbuf, stream);
  scatter_pe_kernel<<<Bsz * T2c, 256, 0, stream>>>(out2, xv2, toff2, t2, T2c);
  run_encoder(out2, Bsz, T2c, ps2, t2, bufA, bufB, bufC, stream);
}
